// MoEFFN_30021821399689
// MI455X (gfx1250) — matrix operand order repaired and checked by reading, not compiled, not run on hardware
//
#include <hip/hip_runtime.h>
#include <hip/hip_bf16.h>
#include <stdint.h>

// ---------------- problem constants ----------------
#define T_TOK 8192      // B*S
#define D_IN  1024      // D
#define D_FF  4096      // DFF
#define N_EXP 8         // E
#define TOPK  2
#define MT_MAX (T_TOK / 64)   // 128 worst-case m-tiles per expert

// ---------------- CDNA5 WMMA types ----------------
typedef __attribute__((ext_vector_type(16))) __bf16 v16bf;
typedef __attribute__((ext_vector_type(8)))  float  v8f;
typedef __attribute__((ext_vector_type(4)))  int    v4i;

// pointer-to-<4 x i32> in global (AS1) / LDS (AS3) address spaces,
// matching the clang prototype of the async-to-LDS builtins.
typedef __attribute__((address_space(1))) v4i* gv4i_p;
typedef __attribute__((address_space(3))) v4i* lv4i_p;

union B128x2 { uint4 q[2]; v16bf v; };

// round-to-nearest-even fp32 -> bf16 (raw bits)
__device__ __forceinline__ unsigned short f2bf(float f) {
  unsigned int u = __float_as_uint(f);
  unsigned int r = 0x7FFFu + ((u >> 16) & 1u);
  return (unsigned short)((u + r) >> 16);
}

// ---------------- CDNA5 async global->LDS copy (ASYNCcnt path) ----------------
__device__ __forceinline__ void async_copy_b128(const unsigned short* g, unsigned short* l) {
#if __has_builtin(__builtin_amdgcn_global_load_async_to_lds_b128)
  // flat LDS address low 32 bits == LDS byte offset (ISA aperture rule)
  __builtin_amdgcn_global_load_async_to_lds_b128(
      (gv4i_p)(uintptr_t)g,
      (lv4i_p)(unsigned int)(uintptr_t)l,
      0, 0);
#else
  // VFLAT async encoding: VDST carries the LDS byte address, VADDR the 64-bit
  // global address (GV mode).
  unsigned int lds_lo = (unsigned int)(uintptr_t)l;
  asm volatile("global_load_async_to_lds_b128 %0, %1, off"
               :: "v"(lds_lo), "v"((const void*)g)
               : "memory");
#endif
}

template <int N>
__device__ __forceinline__ void wait_asynccnt() {
#if __has_builtin(__builtin_amdgcn_s_wait_asynccnt)
  __builtin_amdgcn_s_wait_asynccnt(N);
#else
  asm volatile("s_wait_asynccnt %0" :: "i"(N) : "memory");
#endif
}

// A-fragment (16x32 bf16, MxK): lanes 0-15 hold M=lane, K{0..7,16..23};
// lanes 16-31 hold M=lane-16, K{8..15,24..31}. LDS tile is row-major [M][32].
__device__ __forceinline__ v16bf frag_A(const unsigned short* lds, int mtile, int lane) {
  const int row  = mtile * 16 + (lane & 15);
  const int half = lane >> 4;
  const uint4* p = (const uint4*)(lds + row * 32);
  B128x2 f;
  f.q[0] = p[half];      // bytes half*16       -> K 0..7   / 8..15
  f.q[1] = p[2 + half];  // bytes 32 + half*16  -> K 16..23 / 24..31
  return f.v;
}

// B-fragment (32x16 bf16, KxN): lanes 0-15 hold N=lane, K{0..7,16..23};
// lanes 16-31 hold N=lane-16, K{8..15,24..31}. LDS tile is [N][32] (pre-transposed B).
__device__ __forceinline__ v16bf frag_B(const unsigned short* lds, int ntile, int lane) {
  const int row  = ntile * 16 + (lane & 15);
  const int half = lane >> 4;
  const uint4* p = (const uint4*)(lds + row * 32);
  B128x2 f;
  f.q[0] = p[half];          // K 0..7   / 8..15
  f.q[1] = p[2 + half];      // K 16..23 / 24..31
  return f.v;
}

// ---------------- pre-pass: fp32 -> bf16 elementwise ----------------
__global__ __launch_bounds__(256) void convert_x_kernel(const float* __restrict__ x,
                                                        unsigned short* __restrict__ xb,
                                                        int n4) {
  int i = blockIdx.x * 256 + threadIdx.x;
  if (i >= n4) return;
  float4 v = ((const float4*)x)[i];
  uint2 o;
  o.x = (unsigned)f2bf(v.x) | ((unsigned)f2bf(v.y) << 16);
  o.y = (unsigned)f2bf(v.z) | ((unsigned)f2bf(v.w) << 16);
  ((uint2*)xb)[i] = o;
}

// ---------------- pre-pass: per-expert transpose + fp32->bf16 ----------------
// in: [E][R][C] fp32 ; out: [E][C][R] bf16
__global__ __launch_bounds__(256) void transpose_convert_kernel(const float* __restrict__ in,
                                                                unsigned short* __restrict__ out,
                                                                int R, int C) {
  __shared__ float tile[32][33];
  const size_t per = (size_t)R * C;
  const float* ip = in + (size_t)blockIdx.z * per;
  unsigned short* op = out + (size_t)blockIdx.z * per;
  const int c0 = blockIdx.x * 32;
  const int r0 = blockIdx.y * 32;
  const int tx = threadIdx.x, ty = threadIdx.y;
#pragma unroll
  for (int j = 0; j < 32; j += 8)
    tile[ty + j][tx] = ip[(size_t)(r0 + ty + j) * C + c0 + tx];
  __syncthreads();
#pragma unroll
  for (int j = 0; j < 32; j += 8)
    op[(size_t)(c0 + ty + j) * R + r0 + tx] = f2bf(tile[tx][ty + j]);
}

// ---------------- router: one wave32 per token ----------------
__global__ __launch_bounds__(256) void router_kernel(const float* __restrict__ x,
                                                     const float* __restrict__ rw,
                                                     int* __restrict__ cnt,
                                                     int* __restrict__ tok_list,
                                                     float* __restrict__ gate_list) {
  const int lane  = threadIdx.x & 31;
  const int wave  = threadIdx.x >> 5;
  const int token = blockIdx.x * 8 + wave;
  float acc[N_EXP];
#pragma unroll
  for (int e = 0; e < N_EXP; ++e) acc[e] = 0.0f;

  const float* xr = x + (size_t)token * D_IN;
  for (int d = lane; d < D_IN; d += 32) {
    float xv = xr[d];
    const float4* rwp = (const float4*)(rw + (size_t)d * N_EXP);
    float4 r0 = rwp[0], r1 = rwp[1];
    acc[0] += xv * r0.x; acc[1] += xv * r0.y; acc[2] += xv * r0.z; acc[3] += xv * r0.w;
    acc[4] += xv * r1.x; acc[5] += xv * r1.y; acc[6] += xv * r1.z; acc[7] += xv * r1.w;
  }
#pragma unroll
  for (int off = 16; off > 0; off >>= 1)
#pragma unroll
    for (int e = 0; e < N_EXP; ++e) acc[e] += __shfl_xor(acc[e], off, 32);

  if (lane == 0) {
    float mx = acc[0];
#pragma unroll
    for (int e = 1; e < N_EXP; ++e) mx = fmaxf(mx, acc[e]);
    float p[N_EXP], s = 0.0f;
#pragma unroll
    for (int e = 0; e < N_EXP; ++e) { p[e] = __expf(acc[e] - mx); s += p[e]; }
    float inv = 1.0f / s;
#pragma unroll
    for (int e = 0; e < N_EXP; ++e) p[e] *= inv;
    int i0 = 0;
#pragma unroll
    for (int e = 1; e < N_EXP; ++e) if (p[e] > p[i0]) i0 = e;
    int i1 = (i0 == 0) ? 1 : 0;
#pragma unroll
    for (int e = 0; e < N_EXP; ++e) if (e != i0 && p[e] > p[i1]) i1 = e;
    float w0 = p[i0], w1 = p[i1];
    float ws = 1.0f / (w0 + w1);
    w0 *= ws; w1 *= ws;
    int pos0 = atomicAdd(&cnt[i0], 1);
    tok_list[i0 * T_TOK + pos0]  = token;
    gate_list[i0 * T_TOK + pos0] = w0;
    int pos1 = atomicAdd(&cnt[i1], 1);
    tok_list[i1 * T_TOK + pos1]  = token;
    gate_list[i1 * T_TOK + pos1] = w1;
  }
}

// ---------------- tiny exclusive scan over E=8 ----------------
__global__ void scan_kernel(const int* __restrict__ cnt, int* __restrict__ off) {
  if (threadIdx.x == 0 && blockIdx.x == 0) {
    int a = 0;
#pragma unroll
    for (int e = 0; e < N_EXP; ++e) { off[e] = a; a += cnt[e]; }
    off[N_EXP] = a;
  }
}

// ---------------- pass C: h = gelu(gather(x) @ w_fc[e]) ----------------
// block tile 64(M) x 128(N), K step 32; 8 waves, each 32x32 (2x2 wmma)
// double-buffered LDS filled by async global->LDS (ASYNCcnt) copies
__global__ __launch_bounds__(256) void fc_gelu_kernel(
    const unsigned short* __restrict__ xb,     // [T, D] bf16
    const unsigned short* __restrict__ wfct,   // [E, DFF, D] bf16 (B^T layout)
    const int*  __restrict__ tok_list,         // [E, T]
    const int*  __restrict__ cnt,              // [E]
    const int*  __restrict__ off,              // [E+1]
    unsigned short* __restrict__ h)            // [T*K, DFF] bf16 (compacted)
{
  const int e  = blockIdx.x / MT_MAX;
  const int mt = blockIdx.x % MT_MAX;
  const int count = cnt[e];
  const int m0 = mt * 64;
  if (m0 >= count) return;                 // block-uniform exit
  const int n0 = blockIdx.y * 128;

  __shared__ __align__(16) unsigned short ldsA[2][64 * 32];
  __shared__ __align__(16) unsigned short ldsB[2][128 * 32];

  const int tid  = threadIdx.x;
  const int lane = tid & 31;
  const int wave = tid >> 5;
  const int wm   = wave & 1;   // 2 m-strips of 32
  const int wn   = wave >> 1;  // 4 n-strips of 32

  // A staging: 64 rows x 32 cols, 8 bf16 (16B) per thread.
  // Tail rows are clamped to the last valid token so every wave issues exactly
  // 3 async copies per stage (keeps per-wave ASYNCcnt arithmetic exact);
  // duplicated rows are discarded in the epilogue.
  const int arow = tid >> 2;
  const int acol = (tid & 3) * 8;
  const int mrow = (m0 + arow < count) ? (m0 + arow) : (count - 1);
  const int tokA = tok_list[e * T_TOK + mrow];
  const unsigned short* aptr = xb + (size_t)tokA * D_IN + acol;

  // B staging: 128 rows x 32 cols, two 16B chunks per thread
  const int bidx1 = tid + 256;
  const int brow0 = tid >> 2,   bcol0 = (tid & 3) * 8;
  const int brow1 = bidx1 >> 2, bcol1 = (bidx1 & 3) * 8;
  const unsigned short* bptr0 = wfct + ((size_t)e * D_FF + n0 + brow0) * D_IN + bcol0;
  const unsigned short* bptr1 = wfct + ((size_t)e * D_FF + n0 + brow1) * D_IN + bcol1;

  const int aoff = arow * 32 + acol;
  const int boff0 = brow0 * 32 + bcol0;
  const int boff1 = brow1 * 32 + bcol1;

  auto stage = [&](int buf, int k0) {
    async_copy_b128(aptr  + k0, &ldsA[buf][aoff]);
    async_copy_b128(bptr0 + k0, &ldsB[buf][boff0]);
    async_copy_b128(bptr1 + k0, &ldsB[buf][boff1]);
  };

  v8f acc[2][2] = {};

  stage(0, 0);
  int kb = 0;
  for (int k0 = 0; k0 < D_IN; k0 += 32, kb ^= 1) {
    if (k0 + 32 < D_IN) {
      stage(kb ^ 1, k0 + 32);      // prefetch next tile into other buffer
      wait_asynccnt<3>();          // current tile's 3 copies complete (in-order)
    } else {
      wait_asynccnt<0>();
    }
    __syncthreads();

    v16bf af[2], bf_[2];
    af[0]  = frag_A(ldsA[kb], wm * 2 + 0, lane);
    af[1]  = frag_A(ldsA[kb], wm * 2 + 1, lane);
    bf_[0] = frag_B(ldsB[kb], wn * 2 + 0, lane);
    bf_[1] = frag_B(ldsB[kb], wn * 2 + 1, lane);
#pragma unroll
    for (int mi = 0; mi < 2; ++mi)
#pragma unroll
      for (int ni = 0; ni < 2; ++ni)
        acc[mi][ni] = __builtin_amdgcn_wmma_f32_16x16x32_bf16(
            false, af[mi], false, bf_[ni], (short)0, acc[mi][ni], false, false);
    __syncthreads();
  }

  // epilogue: exact gelu, bf16 store into compacted h rows
  const size_t hbase = (size_t)(off[e] + m0);
  const int half = lane >> 4;
  const int ncol = lane & 15;
#pragma unroll
  for (int mi = 0; mi < 2; ++mi)
#pragma unroll
    for (int ni = 0; ni < 2; ++ni) {
      const int nglob = n0 + (wn * 2 + ni) * 16 + ncol;
#pragma unroll
      for (int r = 0; r < 8; ++r) {
        const int m_local = (wm * 2 + mi) * 16 + r + half * 8;
        if (m0 + m_local < count) {
          float v = acc[mi][ni][r];
          float g = 0.5f * v * (1.0f + erff(v * 0.70710678118654752f));
          h[(hbase + m_local) * (size_t)D_FF + nglob] = f2bf(g);
        }
      }
    }
}

// ---------------- pass D: out[tok] += gate * (h @ w_proj[e]) ----------------
__global__ __launch_bounds__(256) void proj_scatter_kernel(
    const unsigned short* __restrict__ h,       // [T*K, DFF] bf16
    const unsigned short* __restrict__ wprojt,  // [E, D, DFF] bf16 (B^T layout)
    const int*   __restrict__ tok_list,         // [E, T]
    const float* __restrict__ gate_list,        // [E, T]
    const int*   __restrict__ cnt,
    const int*   __restrict__ off,
    float* __restrict__ out)                    // [T, D] fp32 (zeroed)
{
  const int e  = blockIdx.x / MT_MAX;
  const int mt = blockIdx.x % MT_MAX;
  const int count = cnt[e];
  const int m0 = mt * 64;
  if (m0 >= count) return;
  const int n0 = blockIdx.y * 128;

  __shared__ __align__(16) unsigned short ldsA[2][64 * 32];
  __shared__ __align__(16) unsigned short ldsB[2][128 * 32];

  const int tid  = threadIdx.x;
  const int lane = tid & 31;
  const int wave = tid >> 5;
  const int wm   = wave & 1;
  const int wn   = wave >> 1;

  const int arow = tid >> 2;
  const int acol = (tid & 3) * 8;
  const int mrow = (m0 + arow < count) ? (m0 + arow) : (count - 1);
  const unsigned short* aptr = h + (size_t)(off[e] + mrow) * D_FF + acol;

  const int bidx1 = tid + 256;
  const int brow0 = tid >> 2,   bcol0 = (tid & 3) * 8;
  const int brow1 = bidx1 >> 2, bcol1 = (bidx1 & 3) * 8;
  const unsigned short* bptr0 = wprojt + ((size_t)e * D_IN + n0 + brow0) * D_FF + bcol0;
  const unsigned short* bptr1 = wprojt + ((size_t)e * D_IN + n0 + brow1) * D_FF + bcol1;

  const int aoff = arow * 32 + acol;
  const int boff0 = brow0 * 32 + bcol0;
  const int boff1 = brow1 * 32 + bcol1;

  auto stage = [&](int buf, int k0) {
    async_copy_b128(aptr  + k0, &ldsA[buf][aoff]);
    async_copy_b128(bptr0 + k0, &ldsB[buf][boff0]);
    async_copy_b128(bptr1 + k0, &ldsB[buf][boff1]);
  };

  v8f acc[2][2] = {};

  stage(0, 0);
  int kb = 0;
  for (int k0 = 0; k0 < D_FF; k0 += 32, kb ^= 1) {
    if (k0 + 32 < D_FF) {
      stage(kb ^ 1, k0 + 32);
      wait_asynccnt<3>();
    } else {
      wait_asynccnt<0>();
    }
    __syncthreads();

    v16bf af[2], bf_[2];
    af[0]  = frag_A(ldsA[kb], wm * 2 + 0, lane);
    af[1]  = frag_A(ldsA[kb], wm * 2 + 1, lane);
    bf_[0] = frag_B(ldsB[kb], wn * 2 + 0, lane);
    bf_[1] = frag_B(ldsB[kb], wn * 2 + 1, lane);
#pragma unroll
    for (int mi = 0; mi < 2; ++mi)
#pragma unroll
      for (int ni = 0; ni < 2; ++ni)
        acc[mi][ni] = __builtin_amdgcn_wmma_f32_16x16x32_bf16(
            false, af[mi], false, bf_[ni], (short)0, acc[mi][ni], false, false);
    __syncthreads();
  }

  const int half = lane >> 4;
  const int ncol = lane & 15;
#pragma unroll
  for (int mi = 0; mi < 2; ++mi)
#pragma unroll
    for (int ni = 0; ni < 2; ++ni) {
      const int nglob = n0 + (wn * 2 + ni) * 16 + ncol;
#pragma unroll
      for (int r = 0; r < 8; ++r) {
        const int m_local = (wm * 2 + mi) * 16 + r + half * 8;
        if (m0 + m_local < count) {
          const int   tok = tok_list[e * T_TOK + m0 + m_local];
          const float g   = gate_list[e * T_TOK + m0 + m_local];
          atomicAdd(out + (size_t)tok * D_IN + nglob, g * acc[mi][ni][r]);
        }
      }
    }
}

// ---------------- host side ----------------
extern "C" void kernel_launch(void* const* d_in, const int* in_sizes, int n_in,
                              void* d_out, int out_size, void* d_ws, size_t ws_size,
                              hipStream_t stream) {
  (void)in_sizes; (void)n_in; (void)out_size; (void)ws_size;
  const float* x     = (const float*)d_in[0];   // [B,S,D]
  const float* rw    = (const float*)d_in[1];   // [D,E]
  const float* wfc   = (const float*)d_in[2];   // [E,D,DFF]
  const float* wproj = (const float*)d_in[3];   // [E,DFF,D]
  float* out = (float*)d_out;                   // [B,S,D]

  // workspace carve-up (all regions 256B aligned by construction)
  char* w = (char*)d_ws;
  unsigned short* xb     = (unsigned short*)w; w += (size_t)T_TOK * D_IN * 2;          // 16 MB
  unsigned short* wfct   = (unsigned short*)w; w += (size_t)N_EXP * D_FF * D_IN * 2;   // 64 MB
  unsigned short* wprojt = (unsigned short*)w; w += (size_t)N_EXP * D_IN * D_FF * 2;   // 64 MB
  int*   tok_list  = (int*)w;   w += (size_t)N_EXP * T_TOK * 4;                        // 256 KB
  float* gate_list = (float*)w; w += (size_t)N_EXP * T_TOK * 4;                        // 256 KB
  int*   cnt = (int*)w; w += 256;
  int*   off = (int*)w; w += 256;
  unsigned short* hbuf = (unsigned short*)w;    // [T*K, DFF] bf16 = 128 MB

  (void)hipMemsetAsync(cnt, 0, 256, stream);
  (void)hipMemsetAsync(out, 0, (size_t)T_TOK * D_IN * sizeof(float), stream);

  convert_x_kernel<<<(T_TOK * D_IN / 4 + 255) / 256, 256, 0, stream>>>(x, xb, T_TOK * D_IN / 4);

  dim3 tb(32, 8);
  transpose_convert_kernel<<<dim3(D_FF / 32, D_IN / 32, N_EXP), tb, 0, stream>>>(wfc, wfct, D_IN, D_FF);
  transpose_convert_kernel<<<dim3(D_IN / 32, D_FF / 32, N_EXP), tb, 0, stream>>>(wproj, wprojt, D_FF, D_IN);

  router_kernel<<<T_TOK / 8, 256, 0, stream>>>(x, rw, cnt, tok_list, gate_list);
  scan_kernel<<<1, 1, 0, stream>>>(cnt, off);

  fc_gelu_kernel<<<dim3(N_EXP * MT_MAX, D_FF / 128), 256, 0, stream>>>(
      xb, wfct, tok_list, cnt, off, hbuf);
  proj_scatter_kernel<<<dim3(N_EXP * MT_MAX, D_IN / 128), 256, 0, stream>>>(
      hbuf, wprojt, tok_list, gate_list, cnt, off, out);
}
